// CausalSelfAttention_6416681140598
// MI455X (gfx1250) — compile-verified, matrix operand order checked by reading, not run on hardware
//
#include <hip/hip_runtime.h>

typedef __attribute__((ext_vector_type(16))) _Float16 v16h;
typedef __attribute__((ext_vector_type(8)))  _Float16 v8h;
typedef __attribute__((ext_vector_type(8)))  float    v8f;

union V16 { v16h v; v8h h[2]; };

#define WMMA_F16(a, b, c) \
  __builtin_amdgcn_wmma_f32_16x16x32_f16(false, (a), false, (b), (short)0, (c), false, false)

// Flip to 0 if the async-to-LDS mnemonic is rejected by the assembler.
#define ASYNC_LDS 1

constexpr int Bn   = 4;
constexpr int T    = 2048;
constexpr int Cdim = 1024;
constexpr int H    = 16;
constexpr int Dh   = 64;
constexpr int Mtot = Bn * T;          // 8192 token rows
constexpr int LDSP = 40;              // padded LDS row stride in halfs (conflict-free frag reads)

__device__ inline v8f vzero8() {
  v8f z;
#pragma unroll
  for (int i = 0; i < 8; ++i) z[i] = 0.0f;
  return z;
}

// 16B global -> LDS copy. Async path uses the CDNA5 ASYNCcnt-tracked direct
// LDS write (GLOBAL_LOAD_ASYNC_TO_LDS_B128); generic->LDS pointers truncate to
// the 32-bit LDS offset per the ISA aperture rule, which is what VDST wants.
__device__ inline void copy16_g2l(const _Float16* __restrict__ gsrc, _Float16* ldst) {
#if ASYNC_LDS
  unsigned loff = (unsigned)(uintptr_t)ldst;
  unsigned long long ga = (unsigned long long)(uintptr_t)gsrc;
  asm volatile("global_load_async_to_lds_b128 %0, %1, off"
               :: "v"(loff), "v"(ga) : "memory");
#else
  *(v8h*)ldst = *(const v8h*)gsrc;
#endif
}

__device__ inline void wait_async_lds() {
#if ASYNC_LDS
  asm volatile("s_wait_asynccnt 0x0" ::: "memory");
#endif
}

// ---------------------------------------------------------------------------
// f32 -> f16 elementwise convert
// ---------------------------------------------------------------------------
__global__ void k_cvt_f16(const float* __restrict__ in, _Float16* __restrict__ out, int n) {
  int i = blockIdx.x * blockDim.x + threadIdx.x;
  if (i < n) out[i] = (_Float16)in[i];
}

// in: [rows(K)][cols(N)] f32, out: [cols(N)][rows(K)] f16  (B stored column-major for WMMA)
__global__ void k_transpose_cvt(const float* __restrict__ in, _Float16* __restrict__ out,
                                int rows, int cols) {
  int i = blockIdx.x * blockDim.x + threadIdx.x;
  if (i < rows * cols) {
    int n = i / rows;
    int k = i - n * rows;
    out[i] = (_Float16)in[(size_t)k * cols + n];
  }
}

// ---------------------------------------------------------------------------
// GEMM mainloop: 128Mx128N block tile, 256 threads (8 waves), BLK_K=32,
// double-buffered LDS with async global->LDS staging. Wave (w&3, w>>2) owns a
// 32Mx64N tile: 2 A-frags x 4 B-frags = 8 WMMAs per K-step, B-frag reused
// across both M-frags. A: [M][K] f16 row-major; Bt: [N][K] f16 (B col-major).
// ---------------------------------------------------------------------------
constexpr int TILE_HALFS = 128 * LDSP;     // one 128x32 tile (padded)

__device__ inline void stage_tile(const _Float16* __restrict__ G, int ldk,
                                  int row0, int k0, _Float16* S) {
  const int tid = threadIdx.x;
#pragma unroll
  for (int c = tid; c < 512; c += 256) {
    const int row = c >> 2;
    const int kc  = (c & 3) * 8;
    copy16_g2l(&G[(size_t)(row0 + row) * ldk + k0 + kc], &S[row * LDSP + kc]);
  }
}

__device__ inline void gemm_tile_128x128(const _Float16* __restrict__ A,
                                         const _Float16* __restrict__ Bt,
                                         int K, int m0, int n0,
                                         _Float16* sA, _Float16* sB,   // [2][TILE_HALFS] each
                                         v8f acc[2][4]) {
  const int lane = threadIdx.x & 31;
  const int w    = threadIdx.x >> 5;
  const int lm   = lane & 15;
  const int g    = lane >> 4;
  const int wm   = (w & 3) * 32;
  const int wn   = (w >> 2) * 64;

#pragma unroll
  for (int mi = 0; mi < 2; ++mi)
#pragma unroll
    for (int nt = 0; nt < 4; ++nt) acc[mi][nt] = vzero8();

  const int nIter = K / 32;
  stage_tile(A,  K, m0, 0, sA);
  stage_tile(Bt, K, n0, 0, sB);
  wait_async_lds();
  __syncthreads();

  for (int it = 0; it < nIter; ++it) {
    const int cur = (it & 1) * TILE_HALFS;
    const int nxt = ((it + 1) & 1) * TILE_HALFS;
    if (it + 1 < nIter) {                       // prefetch next K-tile (async)
      stage_tile(A,  K, m0, (it + 1) * 32, sA + nxt);
      stage_tile(Bt, K, n0, (it + 1) * 32, sB + nxt);
    }

    V16 a0, a1;
    a0.h[0] = *(const v8h*)&sA[cur + (wm + lm) * LDSP + g * 8];
    a0.h[1] = *(const v8h*)&sA[cur + (wm + lm) * LDSP + 16 + g * 8];
    a1.h[0] = *(const v8h*)&sA[cur + (wm + 16 + lm) * LDSP + g * 8];
    a1.h[1] = *(const v8h*)&sA[cur + (wm + 16 + lm) * LDSP + 16 + g * 8];
#pragma unroll
    for (int nt = 0; nt < 4; ++nt) {
      const int nl = wn + nt * 16 + lm;
      V16 bf;
      bf.h[0] = *(const v8h*)&sB[cur + nl * LDSP + g * 8];
      bf.h[1] = *(const v8h*)&sB[cur + nl * LDSP + 16 + g * 8];
      acc[0][nt] = WMMA_F16(a0.v, bf.v, acc[0][nt]);
      acc[1][nt] = WMMA_F16(a1.v, bf.v, acc[1][nt]);
    }

    wait_async_lds();     // next tile landed
    __syncthreads();      // and everyone is done reading the current tile
  }
}

// ---------------------------------------------------------------------------
// QKV GEMM: qkv = x @ W_attn + b_attn, scatter epilogue:
//   Q -> [b,h,t,d] f16 pre-scaled by 1/sqrt(D); K -> [b,h,t,d]; V -> [b,h,d,t]
// Every 16-wide N-subtile lies inside one head (Dh==64, tiles 64-aligned).
// ---------------------------------------------------------------------------
__global__ void __launch_bounds__(256)
k_gemm_qkv(const _Float16* __restrict__ xh, const _Float16* __restrict__ wt,
           const float* __restrict__ bias,
           _Float16* __restrict__ Qs, _Float16* __restrict__ Kh, _Float16* __restrict__ Vt) {
  __shared__ _Float16 sA[2 * TILE_HALFS];
  __shared__ _Float16 sB[2 * TILE_HALFS];
  const int nBlkN = (3 * Cdim) / 128;          // 24
  const int m0 = (blockIdx.x / nBlkN) * 128;
  const int n0 = (blockIdx.x % nBlkN) * 128;

  v8f acc[2][4];
  gemm_tile_128x128(xh, wt, Cdim, m0, n0, sA, sB, acc);

  const int lane = threadIdx.x & 31;
  const int w    = threadIdx.x >> 5;
  const int lm   = lane & 15;
  const int g    = lane >> 4;

#pragma unroll
  for (int mi = 0; mi < 2; ++mi) {
    const int mb = m0 + (w & 3) * 32 + mi * 16;
#pragma unroll
    for (int nt = 0; nt < 4; ++nt) {
      const int n     = n0 + (w >> 2) * 64 + nt * 16;
      const int third = n / Cdim;
      const int c     = n % Cdim;
      const int hh    = c / Dh;
      const int dbase = (c % Dh) + lm;
      const float bsc = bias[n + lm];
      const float scl = (third == 0) ? 0.125f : 1.0f;   // 1/sqrt(64)
#pragma unroll
      for (int r = 0; r < 8; ++r) {
        const int m = mb + r + 8 * g;
        const int b = m >> 11;
        const int t = m & (T - 1);
        const _Float16 hv = (_Float16)((acc[mi][nt][r] + bsc) * scl);
        const size_t bh = (size_t)(b * H + hh);
        if (third == 0)      Qs[(bh * T + t) * Dh + dbase] = hv;
        else if (third == 1) Kh[(bh * T + t) * Dh + dbase] = hv;
        else                 Vt[(bh * Dh + dbase) * T + t] = hv;
      }
    }
  }
}

// ---------------------------------------------------------------------------
// Flash attention: one wave per (b, h, 16-row query tile). Streams 32-key
// blocks up to the causal diagonal. S = Q K^T via WMMA, online softmax with
// wave32 shfl_xor row reductions, P re-shaped D-layout -> A-layout via a
// per-wave padded LDS tile, O += P V via WMMA. Output f16 [b,t,h*D] (= yh).
// ---------------------------------------------------------------------------
__global__ void __launch_bounds__(128)
k_attn(const _Float16* __restrict__ Qs, const _Float16* __restrict__ Kh,
       const _Float16* __restrict__ Vt, _Float16* __restrict__ yh) {
  __shared__ _Float16 pbuf[4][16 * LDSP];
  const int lane = threadIdx.x & 31;
  const int wv   = threadIdx.x >> 5;
  const int lm   = lane & 15;
  const int g    = lane >> 4;

  const int wid     = blockIdx.x * 4 + wv;
  const int itile   = wid & (T / 16 - 1);     // 128 tiles per (b,h)
  const int bh      = wid >> 7;
  const int rowbase = itile * 16;

  const _Float16* Qb = Qs + (size_t)bh * T * Dh;
  const _Float16* Kb = Kh + (size_t)bh * T * Dh;
  const _Float16* Vb = Vt + (size_t)bh * Dh * T;
  _Float16* sp = pbuf[wv];

  // Q A-fragments: 16 rows x 64 d, two K=32 chunks. Contiguous 16B loads.
  V16 qa[2];
  {
    const _Float16* qrow = Qb + (size_t)(rowbase + lm) * Dh;
#pragma unroll
    for (int c = 0; c < 2; ++c) {
      qa[c].h[0] = *(const v8h*)&qrow[c * 32 + g * 8];
      qa[c].h[1] = *(const v8h*)&qrow[c * 32 + 16 + g * 8];
    }
  }

  v8f o[4];
#pragma unroll
  for (int dt = 0; dt < 4; ++dt) o[dt] = vzero8();
  float mrun[8], lrun[8];
#pragma unroll
  for (int r = 0; r < 8; ++r) { mrun[r] = -1e30f; lrun[r] = 0.0f; }

  const int nj = rowbase / 32 + 1;
  for (int j = 0; j < nj; ++j) {
    const int jbase = j * 32;

    // S = Q * K^T for this 16x32 key block (two 16x16 N-tiles).
    v8f s0 = vzero8(), s1 = vzero8();
#pragma unroll
    for (int nt = 0; nt < 2; ++nt) {
      const _Float16* krow = Kb + (size_t)(jbase + nt * 16 + lm) * Dh;
#pragma unroll
      for (int c = 0; c < 2; ++c) {
        V16 kf;
        kf.h[0] = *(const v8h*)&krow[c * 32 + g * 8];
        kf.h[1] = *(const v8h*)&krow[c * 32 + 16 + g * 8];
        if (nt == 0) s0 = WMMA_F16(qa[c].v, kf.v, s0);
        else         s1 = WMMA_F16(qa[c].v, kf.v, s1);
      }
    }

    // Causal mask: only the diagonal block can contain s > t.
    if (j == nj - 1) {
#pragma unroll
      for (int r = 0; r < 8; ++r) {
        const int t = rowbase + r + 8 * g;
        if (jbase + lm      > t) s0[r] = -1e30f;
        if (jbase + 16 + lm > t) s1[r] = -1e30f;
      }
    }

    // Online softmax. Row m for register r lives across the 16 lanes of a
    // half-wave (C/D layout), so shfl_xor masks 1..8 reduce exactly one row.
#pragma unroll
    for (int r = 0; r < 8; ++r) {
      float vmax = fmaxf(s0[r], s1[r]);
#pragma unroll
      for (int off = 1; off < 16; off <<= 1) vmax = fmaxf(vmax, __shfl_xor(vmax, off, 32));
      const float mnew  = fmaxf(mrun[r], vmax);
      const float scale = __expf(mrun[r] - mnew);
      const float p0 = __expf(s0[r] - mnew);
      const float p1 = __expf(s1[r] - mnew);
      float ps = p0 + p1;
#pragma unroll
      for (int off = 1; off < 16; off <<= 1) ps += __shfl_xor(ps, off, 32);
      lrun[r] = lrun[r] * scale + ps;
      mrun[r] = mnew;
#pragma unroll
      for (int dt = 0; dt < 4; ++dt) o[dt][r] *= scale;
      const int row = r + 8 * g;
      sp[row * LDSP + lm]      = (_Float16)p0;
      sp[row * LDSP + 16 + lm] = (_Float16)p1;
    }
    asm volatile("" ::: "memory");   // keep LDS stores before the re-shaped loads

    // P as a 16x32 A-fragment (D-layout -> A-layout via LDS).
    V16 pa;
    pa.h[0] = *(const v8h*)&sp[lm * LDSP + g * 8];
    pa.h[1] = *(const v8h*)&sp[lm * LDSP + 16 + g * 8];

    // O += P * V ; V is [d][t] so B-fragments are contiguous along t.
#pragma unroll
    for (int dt = 0; dt < 4; ++dt) {
      const _Float16* vrow = Vb + (size_t)(dt * 16 + lm) * T + jbase;
      V16 vf;
      vf.h[0] = *(const v8h*)&vrow[g * 8];
      vf.h[1] = *(const v8h*)&vrow[16 + g * 8];
      o[dt] = WMMA_F16(pa.v, vf.v, o[dt]);
    }
  }

  // Normalize and emit f16 rows of y ([b, t, h*D]) for the projection GEMM.
  const int b  = bh >> 4;
  const int hh = bh & (H - 1);
#pragma unroll
  for (int r = 0; r < 8; ++r) {
    const float inv = 1.0f / lrun[r];
    const int trow = rowbase + r + 8 * g;
    const size_t base = ((size_t)(b * T + trow)) * Cdim + hh * Dh;
#pragma unroll
    for (int dt = 0; dt < 4; ++dt)
      yh[base + dt * 16 + lm] = (_Float16)(o[dt][r] * inv);
  }
}

// ---------------------------------------------------------------------------
// Projection GEMM: out = y @ W_proj + b_proj (f32 output)
// ---------------------------------------------------------------------------
__global__ void __launch_bounds__(256)
k_gemm_proj(const _Float16* __restrict__ yh, const _Float16* __restrict__ wt,
            const float* __restrict__ bias, float* __restrict__ out) {
  __shared__ _Float16 sA[2 * TILE_HALFS];
  __shared__ _Float16 sB[2 * TILE_HALFS];
  const int nBlkN = Cdim / 128;                // 8
  const int m0 = (blockIdx.x / nBlkN) * 128;
  const int n0 = (blockIdx.x % nBlkN) * 128;

  v8f acc[2][4];
  gemm_tile_128x128(yh, wt, Cdim, m0, n0, sA, sB, acc);

  const int lane = threadIdx.x & 31;
  const int w    = threadIdx.x >> 5;
  const int lm   = lane & 15;
  const int g    = lane >> 4;

#pragma unroll
  for (int mi = 0; mi < 2; ++mi) {
    const int mb = m0 + (w & 3) * 32 + mi * 16;
#pragma unroll
    for (int nt = 0; nt < 4; ++nt) {
      const int n = n0 + (w >> 2) * 64 + nt * 16 + lm;
      const float bsc = bias[n];
#pragma unroll
      for (int r = 0; r < 8; ++r) {
        const int m = mb + r + 8 * g;
        out[(size_t)m * Cdim + n] = acc[mi][nt][r] + bsc;
      }
    }
  }
}

// ---------------------------------------------------------------------------
// Host side
// ---------------------------------------------------------------------------
extern "C" void kernel_launch(void* const* d_in, const int* in_sizes, int n_in,
                              void* d_out, int out_size, void* d_ws, size_t ws_size,
                              hipStream_t stream) {
  (void)in_sizes; (void)n_in; (void)out_size; (void)ws_size;
  const float* x      = (const float*)d_in[0];
  const float* w_attn = (const float*)d_in[1];
  const float* b_attn = (const float*)d_in[2];
  const float* w_proj = (const float*)d_in[3];
  const float* b_proj = (const float*)d_in[4];
  float* out = (float*)d_out;

  char* ws = (char*)d_ws;
  size_t off = 0;
  auto carve = [&](size_t bytes) {
    void* p = ws + off;
    off += (bytes + 255) & ~(size_t)255;
    return p;
  };
  const size_t tokBytes = (size_t)Mtot * Cdim * sizeof(_Float16);      // 16 MiB
  _Float16* xh      = (_Float16*)carve(tokBytes);   // reused as yh after QKV
  _Float16* wattn_t = (_Float16*)carve((size_t)3 * Cdim * Cdim * sizeof(_Float16));
  _Float16* wproj_t = (_Float16*)carve((size_t)Cdim * Cdim * sizeof(_Float16));
  _Float16* Qs      = (_Float16*)carve((size_t)Bn * H * T * Dh * sizeof(_Float16));
  _Float16* Kh      = (_Float16*)carve((size_t)Bn * H * T * Dh * sizeof(_Float16));
  _Float16* Vt      = (_Float16*)carve((size_t)Bn * H * T * Dh * sizeof(_Float16));
  _Float16* yh      = xh;   // alias: x no longer needed once attention runs

  const int nx = Mtot * Cdim;
  k_cvt_f16<<<nx / 256, 256, 0, stream>>>(x, xh, nx);
  k_transpose_cvt<<<(3 * Cdim * Cdim) / 256, 256, 0, stream>>>(w_attn, wattn_t, Cdim, 3 * Cdim);
  k_transpose_cvt<<<(Cdim * Cdim) / 256, 256, 0, stream>>>(w_proj, wproj_t, Cdim, Cdim);

  k_gemm_qkv<<<(Mtot / 128) * ((3 * Cdim) / 128), 256, 0, stream>>>(xh, wattn_t, b_attn, Qs, Kh, Vt);
  k_attn<<<(Bn * H * (T / 16)) / 4, 128, 0, stream>>>(Qs, Kh, Vt, yh);
  k_gemm_proj<<<(Mtot / 128) * (Cdim / 128), 256, 0, stream>>>(yh, wproj_t, b_proj, out);
}
